// LSTM_86088324481200
// MI455X (gfx1250) — matrix operand order repaired and checked by reading, not compiled, not run on hardware
//
#include <hip/hip_runtime.h>
#include <stdint.h>

typedef __attribute__((ext_vector_type(16))) _Float16 v16h;
typedef __attribute__((ext_vector_type(8)))  float    v8f;

enum : int { T_STEPS = 1024, BATCH = 64, ISIZE = 512, HSIZE = 512, NGATE = 4 };

// ---------------- workspace layout (bytes, all 256-aligned) ----------------
static const size_t OFF_X   = 0;                                     // X f16 [T*B, I]
static const size_t SZ_X    = (size_t)T_STEPS * BATCH * ISIZE * 2;   // 64 MB
static const size_t OFF_WX  = OFF_X + SZ_X;                          // WxT f16 [g][n][k]
static const size_t SZ_WX   = (size_t)NGATE * HSIZE * ISIZE * 2;     // 2 MB
static const size_t OFF_WH  = OFF_WX + SZ_WX;                        // WhT f16 [g][n][k]
static const size_t SZ_WH   = (size_t)NGATE * HSIZE * HSIZE * 2;     // 2 MB
static const size_t OFF_G   = OFF_WH + SZ_WH;                        // gates f16 [t][g][b][h]
static const size_t SZ_G    = (size_t)T_STEPS * NGATE * BATCH * HSIZE * 2; // 256 MB
static const size_t OFF_H   = OFF_G + SZ_G;                          // h f16 double buffer [2][B*H]
static const size_t SZ_H    = (size_t)2 * BATCH * HSIZE * 2;         // 128 KB
static const size_t OFF_CNT = OFF_H + SZ_H;                          // grid-sync counter

// ---------------- helpers ----------------
struct B32x2 { uint4 lo, hi; };

// B fragment (32x16 f16): lane holds col n=lane&15, K = (lane>>4)*16 .. +15 -> 32 contiguous bytes
__device__ __forceinline__ v16h load_b_frag(const _Float16* p) {
  B32x2 s;
  s.lo = *reinterpret_cast<const uint4*>(p);
  s.hi = *reinterpret_cast<const uint4*>(p + 8);
  return __builtin_bit_cast(v16h, s);
}

// A fragment (16x32 f16): lane holds row m=lane&15; elems 0..15 -> K=khalf..+15 (contiguous)
__device__ __forceinline__ v16h load_a_frag(const _Float16* row, int khalf) {
  B32x2 s;
  s.lo = *reinterpret_cast<const uint4*>(row + khalf);
  s.hi = *reinterpret_cast<const uint4*>(row + khalf + 8);
  return __builtin_bit_cast(v16h, s);
}

__device__ __forceinline__ v8f wmma_f16(v16h a, v16h b, v8f c) {
  return __builtin_amdgcn_wmma_f32_16x16x32_f16(false, a, false, b, (short)0, c, false, false);
}

__device__ __forceinline__ float sigmoidf_fast(float x) {
  return 1.0f / (1.0f + __expf(-x));
}

// ---------------- prep kernels ----------------
__global__ void k_convert_f16(const float* __restrict__ src, _Float16* __restrict__ dst, int n) {
  for (int i = blockIdx.x * blockDim.x + threadIdx.x; i < n; i += gridDim.x * blockDim.x)
    dst[i] = (_Float16)src[i];
}

// dst[n*512 + k] = f16(src[k*512 + n])  (512x512 transpose-convert)
__global__ void k_transpose_w(const float* __restrict__ src, _Float16* __restrict__ dst) {
  const int total = 512 * 512;
  for (int i = blockIdx.x * blockDim.x + threadIdx.x; i < total; i += gridDim.x * blockDim.x) {
    int n = i >> 9, k = i & 511;
    dst[i] = (_Float16)src[k * 512 + n];
  }
}

__global__ void k_zero_counter(unsigned int* c) { *c = 0u; }

// ---------------- phase 1: gates[t][g][b][h] = X @ Wx[g] + bias[g] ----------------
// block = one (row-tile, gate): 8 waves cover all 512 N columns; A tile staged in LDS.
__global__ void __launch_bounds__(256)
k_xproj(const _Float16* __restrict__ X, const _Float16* __restrict__ WxT,
        const float* __restrict__ b_i, const float* __restrict__ b_f,
        const float* __restrict__ b_o, const float* __restrict__ b_u,
        _Float16* __restrict__ gates) {
  __shared__ _Float16 lda[16 * ISIZE];             // 16 KB: shared A tile (16 rows x 512 K)

  const int lane = threadIdx.x & 31;
  const int l15  = lane & 15;
  const int lhi  = lane >> 4;
  const int w    = threadIdx.x >> 5;               // wave in block: N block 0..7
  const int g    = blockIdx.x & 3;                 // gate
  const int mt   = blockIdx.x >> 2;                // 0..4095 row tiles
  const int mbase = mt * 16;
  const int ncb   = w * 64;

  // cooperative stage of A tile into LDS (1024 uint4)
  {
    const uint4* src = reinterpret_cast<const uint4*>(X + (size_t)mbase * ISIZE);
    uint4* dst = reinterpret_cast<uint4*>(lda);
#pragma unroll
    for (int i = 0; i < 4; ++i)
      dst[threadIdx.x + 256 * i] = src[threadIdx.x + 256 * i];
  }
  __syncthreads();

  const _Float16* arow = lda + (size_t)l15 * ISIZE;
  const _Float16* wg   = WxT + (size_t)g * HSIZE * ISIZE;
  const _Float16* wp0  = wg + (size_t)(ncb +  0 + l15) * ISIZE;
  const _Float16* wp1  = wg + (size_t)(ncb + 16 + l15) * ISIZE;
  const _Float16* wp2  = wg + (size_t)(ncb + 32 + l15) * ISIZE;
  const _Float16* wp3  = wg + (size_t)(ncb + 48 + l15) * ISIZE;

  v8f acc[4] = {{}, {}, {}, {}};
  // software pipeline: B fragments (global) prefetched one K-chunk ahead
  v16h b0 = load_b_frag(wp0 + lhi * 16);
  v16h b1 = load_b_frag(wp1 + lhi * 16);
  v16h b2 = load_b_frag(wp2 + lhi * 16);
  v16h b3 = load_b_frag(wp3 + lhi * 16);
#pragma unroll
  for (int kc = 0; kc < ISIZE; kc += 32) {
    v16h a = load_a_frag(arow, kc + lhi * 16);     // LDS, low latency
    v16h n0 = b0, n1 = b1, n2 = b2, n3 = b3;
    if (kc + 32 < ISIZE) {
      const int kbn = kc + 32 + lhi * 16;
      n0 = load_b_frag(wp0 + kbn);
      n1 = load_b_frag(wp1 + kbn);
      n2 = load_b_frag(wp2 + kbn);
      n3 = load_b_frag(wp3 + kbn);
    }
    acc[0] = wmma_f16(a, b0, acc[0]);
    acc[1] = wmma_f16(a, b1, acc[1]);
    acc[2] = wmma_f16(a, b2, acc[2]);
    acc[3] = wmma_f16(a, b3, acc[3]);
    b0 = n0; b1 = n1; b2 = n2; b3 = n3;
  }

  const float* bias = (g == 0) ? b_i : (g == 1) ? b_f : (g == 2) ? b_o : b_u;
  const int t     = mbase >> 6;                    // 64 rows per timestep; 16-row tiles never cross t
  const int brow0 = mbase & 63;
  _Float16* gout = gates + (size_t)((t * 4 + g) * 64) * HSIZE;
#pragma unroll
  for (int nt = 0; nt < 4; ++nt) {
    const int n  = ncb + nt * 16 + l15;
    const float bv = bias[n];
#pragma unroll
    for (int v = 0; v < 8; ++v) {
      const int b = brow0 + v + 8 * lhi;           // D layout: M = v + 8*(lane>=16)
      gout[(size_t)b * HSIZE + n] = (_Float16)(acc[nt][v] + bv);
    }
  }
}

// ---------------- phase 2: persistent recurrent scan ----------------
// 16 blocks x 8 waves = 128 waves = 4 (batch tiles) x 32 (hidden tiles). Each block keeps its
// 128 KB slice of Wh (4 gates x 32 columns x 512 K, f16) resident in LDS for all 1024 steps.
// Cell state c lives in registers for the whole sequence.
__global__ void __launch_bounds__(256)
k_scan(const _Float16* __restrict__ WhT, const _Float16* __restrict__ gates,
       const float* __restrict__ c0, _Float16* __restrict__ hbuf,
       unsigned int* __restrict__ counter, float* __restrict__ out) {
  extern __shared__ _Float16 ldsw[];               // [4][32][512] f16 = 128 KB (dynamic)

  const int lane = threadIdx.x & 31;
  const int l15  = lane & 15;
  const int lhi  = lane >> 4;
  const int w    = threadIdx.x >> 5;               // 0..7
  const int wid  = blockIdx.x * 8 + w;             // 0..127
  const int mbase = (wid & 3) * 16;                // batch tile
  const int ntile = wid >> 2;                      // 0..31
  const int ncol  = ntile * 16 + l15;
  const int blockn0 = blockIdx.x * 32;             // first hidden column owned by block

  // ---- stage this block's Wh slice into LDS: 8192 uint4 ----
  {
    const uint4* src = reinterpret_cast<const uint4*>(WhT);
    uint4* dst = reinterpret_cast<uint4*>(ldsw);
    for (int i = threadIdx.x; i < 8192; i += 256) {
      const int row = i >> 6;                      // g*32 + nloc
      const int kq  = i & 63;
      const int g    = row >> 5;
      const int nloc = row & 31;
      dst[i] = src[((size_t)g * HSIZE + blockn0 + nloc) * 64 + kq];
    }
  }
  __syncthreads();

  const int nloc0 = (ncol - blockn0);              // 0..31, per-lane column within block slice
  const _Float16* p0 = ldsw + ((size_t)(0 * 32 + nloc0)) * HSIZE;
  const _Float16* p1 = ldsw + ((size_t)(1 * 32 + nloc0)) * HSIZE;
  const _Float16* p2 = ldsw + ((size_t)(2 * 32 + nloc0)) * HSIZE;
  const _Float16* p3 = ldsw + ((size_t)(3 * 32 + nloc0)) * HSIZE;

  float c[8], hlast[8];
#pragma unroll
  for (int v = 0; v < 8; ++v) {
    c[v] = c0[(size_t)(mbase + v + 8 * lhi) * HSIZE + ncol];
    hlast[v] = 0.0f;
  }

  const unsigned nwg = gridDim.x;

  for (int t = 0; t < T_STEPS; ++t) {
    const _Float16* hprev = hbuf + (size_t)(t & 1) * BATCH * HSIZE;
    _Float16*       hnext = hbuf + (size_t)((t + 1) & 1) * BATCH * HSIZE;
    const _Float16* arow  = hprev + (size_t)(mbase + l15) * HSIZE;
    const _Float16* gt = gates + (size_t)t * NGATE * BATCH * HSIZE;

    // hoist gate-preactivation loads: latency hides behind the 64 WMMAs below
    float gi[8], gf[8], go[8], gu[8];
#pragma unroll
    for (int v = 0; v < 8; ++v) {
      const size_t rb = (size_t)(mbase + v + 8 * lhi) * HSIZE + ncol;
      gi[v] = (float)gt[0 * BATCH * HSIZE + rb];
      gf[v] = (float)gt[1 * BATCH * HSIZE + rb];
      go[v] = (float)gt[2 * BATCH * HSIZE + rb];
      gu[v] = (float)gt[3 * BATCH * HSIZE + rb];
    }

    v8f ai = {}, af = {}, ao = {}, au = {};
    // software pipeline: A fragment (global h) prefetched one K-chunk ahead; B from LDS
    v16h a = load_a_frag(arow, lhi * 16);
#pragma unroll
    for (int kc = 0; kc < HSIZE; kc += 32) {
      v16h an = a;
      if (kc + 32 < HSIZE) an = load_a_frag(arow, kc + 32 + lhi * 16);
      const int kb = kc + lhi * 16;
      v16h b0 = load_b_frag(p0 + kb);
      v16h b1 = load_b_frag(p1 + kb);
      v16h b2 = load_b_frag(p2 + kb);
      v16h b3 = load_b_frag(p3 + kb);
      ai = wmma_f16(a, b0, ai);
      af = wmma_f16(a, b1, af);
      ao = wmma_f16(a, b2, ao);
      au = wmma_f16(a, b3, au);
      a = an;
    }

#pragma unroll
    for (int v = 0; v < 8; ++v) {
      const int b = mbase + v + 8 * lhi;
      const size_t rb = (size_t)b * HSIZE + ncol;
      const float ig = sigmoidf_fast(ai[v] + gi[v]);
      const float fg = sigmoidf_fast(af[v] + gf[v]);
      const float og = sigmoidf_fast(ao[v] + go[v]);
      const float ug = tanhf(au[v] + gu[v]);
      const float cv = fg * c[v] + ig * ug;
      c[v] = cv;
      const float hv = og * tanhf(cv);
      hlast[v] = hv;
      out[(size_t)t * BATCH * HSIZE + rb] = hv;
      hnext[rb] = (_Float16)hv;                    // single v_cvt_f16_f32
    }

    // ---- grid-wide barrier (release writes of h_t, acquire before reading it) ----
    __threadfence();
    __syncthreads();
    if (threadIdx.x == 0) {
      __hip_atomic_fetch_add(counter, 1u, __ATOMIC_ACQ_REL, __HIP_MEMORY_SCOPE_AGENT);
      const unsigned target = (unsigned)(t + 1) * nwg;
      while (__hip_atomic_load(counter, __ATOMIC_ACQUIRE, __HIP_MEMORY_SCOPE_AGENT) < target)
        __builtin_amdgcn_s_sleep(1);
    }
    __syncthreads();
  }

  float* hT = out + (size_t)T_STEPS * BATCH * HSIZE;
  float* cT = hT + (size_t)BATCH * HSIZE;
#pragma unroll
  for (int v = 0; v < 8; ++v) {
    const int b = mbase + v + 8 * lhi;
    const size_t rb = (size_t)b * HSIZE + ncol;
    hT[rb] = hlast[v];
    cT[rb] = c[v];
  }
}

// ---------------- host ----------------
extern "C" void kernel_launch(void* const* d_in, const int* in_sizes, int n_in,
                              void* d_out, int out_size, void* d_ws, size_t ws_size,
                              hipStream_t stream) {
  (void)in_sizes; (void)n_in; (void)out_size; (void)ws_size;

  const float* x    = (const float*)d_in[0];
  const float* h0   = (const float*)d_in[1];
  const float* c0   = (const float*)d_in[2];
  const float* W_xi = (const float*)d_in[3];
  const float* W_hi = (const float*)d_in[4];
  const float* W_xf = (const float*)d_in[5];
  const float* W_hf = (const float*)d_in[6];
  const float* W_xu = (const float*)d_in[7];
  const float* W_hu = (const float*)d_in[8];
  const float* W_xo = (const float*)d_in[9];
  const float* W_ho = (const float*)d_in[10];
  const float* b_i  = (const float*)d_in[11];
  const float* b_f  = (const float*)d_in[12];
  const float* b_o  = (const float*)d_in[13];
  const float* b_u  = (const float*)d_in[14];

  char* ws = (char*)d_ws;
  _Float16*     Xb    = (_Float16*)(ws + OFF_X);
  _Float16*     WxT   = (_Float16*)(ws + OFF_WX);
  _Float16*     WhT   = (_Float16*)(ws + OFF_WH);
  _Float16*     gates = (_Float16*)(ws + OFF_G);
  _Float16*     hbuf  = (_Float16*)(ws + OFF_H);
  unsigned int* cnt   = (unsigned int*)(ws + OFF_CNT);

  // prep: f16 conversions, weight transposes, h0 into buffer 0, zero the barrier counter
  k_convert_f16<<<4096, 256, 0, stream>>>(x, Xb, T_STEPS * BATCH * ISIZE);
  k_convert_f16<<<128, 256, 0, stream>>>(h0, hbuf, BATCH * HSIZE);
  k_zero_counter<<<1, 1, 0, stream>>>(cnt);

  const float* wx[4] = {W_xi, W_xf, W_xo, W_xu};   // gate order: i, f, o, u
  const float* wh[4] = {W_hi, W_hf, W_ho, W_hu};
  for (int g = 0; g < 4; ++g) {
    k_transpose_w<<<512, 256, 0, stream>>>(wx[g], WxT + (size_t)g * 512 * 512);
    k_transpose_w<<<512, 256, 0, stream>>>(wh[g], WhT + (size_t)g * 512 * 512);
  }

  // phase 1: all input-side projections as one big WMMA GEMM (16384 blocks)
  k_xproj<<<16384, 256, 0, stream>>>(Xb, WxT, b_i, b_f, b_o, b_u, gates);

  // phase 2: persistent recurrent scan, one launch, Wh resident in LDS (128 KB dynamic)
  k_scan<<<16, 256, 128 * 1024, stream>>>(WhT, gates, c0, hbuf, cnt, (float*)d_out);
}
